// VariationHierarchicalCSNet_48223892799920
// MI455X (gfx1250) — matrix operand order repaired and checked by reading, not compiled, not run on hardware
//
#include <hip/hip_runtime.h>
#include <hip/hip_bf16.h>

typedef __attribute__((ext_vector_type(16))) _Float16 v16h;
typedef __attribute__((ext_vector_type(8)))  float    v8f;
typedef __attribute__((ext_vector_type(4)))  unsigned int u32x4;
typedef __attribute__((ext_vector_type(8)))  int      i32x8;
typedef __attribute__((ext_vector_type(4)))  int      i32x4;

#define G    8
#define SC   13
#define OUTC 1024
#define H    96
#define W    96
#define HP   98
#define WP   98

#define NT    3          // N tiles (16 px each) per wave -> 48 output cols per block
#define TCOLS (NT*16+2)  // 50 padded input cols staged in LDS
#define TROWS 3          // 3 padded input rows staged in LDS

// ---------------- workspace layout (bytes, 256-aligned) ----------------
static constexpr size_t OFF_S     = 0;          // 8*13*9 f32          = 3744
static constexpr size_t OFF_PS    = 3840;       // 8*98*98 f32         = 307328
static constexpr size_t OFF_FUSED = 311296;     // 98*98*1024 f16      = 19668992
static constexpr size_t OFF_ACTA  = 19980288;   // 98*98*64 f16        = 1229312
static constexpr size_t OFF_ACTB  = 21209600;
static constexpr size_t OFF_ACTC  = 22438912;
static constexpr size_t OFF_W2P   = 23668224;   // 64*1024*9 f16       = 1179648
static constexpr size_t OFF_W3P   = 24847872;   // 64*64*9 f16         = 73728
static constexpr size_t OFF_W4P   = 24921600;

// ---------------- zero scratch (padding borders must be zero) ----------------
__global__ void k_zero(uint4* p, size_t n16) {
  size_t i = (size_t)blockIdx.x * blockDim.x + threadIdx.x;
  size_t stride = (size_t)gridDim.x * blockDim.x;
  uint4 z; z.x = z.y = z.z = z.w = 0u;
  for (; i < n16; i += stride) p[i] = z;
}

// ---------------- sampling: stride-32 conv, s[g*13+k][3*3] ----------------
__global__ void k_sampling(const float* __restrict__ x, const float* __restrict__ Ws,
                           float* __restrict__ s) {
  int t = blockIdx.x * blockDim.x + threadIdx.x;      // 936 outputs
  if (t >= G * SC * 9) return;
  int ij = t % 9, gk = t / 9;
  int i = ij / 3, j = ij % 3;
  const float* wp = Ws + (size_t)gk * 1024;
  const float* xp = x + (i * 32) * W + j * 32;
  float acc = 0.f;
  for (int py = 0; py < 32; ++py)
    for (int px = 0; px < 32; ++px)
      acc += xp[py * W + px] * wp[py * 32 + px];
  s[t] = acc;
}

// ---------------- head 1x1 conv + pixel-shuffle into padded ps ----------------
__global__ void k_head(const float* __restrict__ s, const float* __restrict__ Wup,
                       const float* __restrict__ bup, float* __restrict__ psPad) {
  int t = blockIdx.x * blockDim.x + threadIdx.x;      // 8*96*96
  if (t >= G * H * W) return;
  int x = t % W; int r = t / W; int y = r % H; int g = r / H;
  int ch = (y % 32) * 32 + (x % 32);                  // pixel-shuffle channel
  int i = y / 32, j = x / 32;
  float acc = bup[ch];
  const float* sp = s + g * SC * 9 + i * 3 + j;
  const float* wp = Wup + ch * SC;
  #pragma unroll
  for (int k = 0; k < SC; ++k) acc += wp[k] * sp[k * 9];
  psPad[((size_t)g * HP + (y + 1)) * WP + (x + 1)] = acc;
}

// ---------------- conv1(1->1024) + PReLU + sum over groups -> fused NHWC f16 ----
__global__ void k_conv1_fuse(const float* __restrict__ psPad, const float* __restrict__ Wc1,
                             const float* __restrict__ bc1, const float* __restrict__ ac1,
                             _Float16* __restrict__ fused) {
  int t = blockIdx.x * blockDim.x + threadIdx.x;      // 96*96*1024
  if (t >= H * W * OUTC) return;
  int c = t % OUTC; int pix = t / OUTC;
  int x = pix % W, y = pix / W;
  float w[9];
  #pragma unroll
  for (int k = 0; k < 9; ++k) w[k] = Wc1[c * 9 + k];
  float b = bc1[c], a = ac1[0];
  float acc = 0.f;
  for (int g = 0; g < G; ++g) {
    const float* pp = psPad + ((size_t)g * HP + y) * WP + x;  // padded: rows y..y+2
    float v = b;
    #pragma unroll
    for (int dy = 0; dy < 3; ++dy)
      #pragma unroll
      for (int dx = 0; dx < 3; ++dx)
        v += w[dy * 3 + dx] * pp[dy * WP + dx];
    acc += (v >= 0.f) ? v : a * v;
  }
  fused[((size_t)(y + 1) * WP + (x + 1)) * OUTC + c] = (_Float16)acc;
}

// ---------------- pack OIHW f32 weights into WMMA A-fragment lane order -------
// consumer index: (((kc*9 + tap)*4 + mtile)*32 + lane)*16 + e     (kc OUTER)
// A 16x32 f16 order used here (B matches): lane = (half<<4)|m ; element e -> K = half*16+e
__global__ void k_pack_w(const float* __restrict__ Wsrc, _Float16* __restrict__ dst, int CIN) {
  int KC = CIN / 32;
  int t = blockIdx.x * blockDim.x + threadIdx.x;
  if (t >= KC * 9 * 4 * 32) return;
  int lane = t % 32; int r = t / 32;
  int mt = r % 4; r /= 4;
  int tap = r % 9; int kc = r / 9;
  int m = mt * 16 + (lane & 15);
  int half = lane >> 4;
  _Float16* out = dst + (size_t)t * 16;
  #pragma unroll
  for (int e = 0; e < 16; ++e) {
    int k  = half * 16 + e;
    int ci = kc * 32 + k;
    out[e] = (_Float16)Wsrc[((size_t)m * CIN + ci) * 9 + tap];
  }
}

// ---------------- TDM: DMA one 32ch x 50col x 3row NHWC halo tile into LDS ----
// D# per CDNA5 ISA ch.8: group0 = {count/type, lds_addr, global_addr},
// group1 = {data_size, tensor_dim0/1, tile_dim0/1/2, dim0/1 strides}, group2 = dim2.
__device__ __forceinline__ void tdm_load_tile(const _Float16* gbase, unsigned lds_addr,
                                              int cin) {
  unsigned long long ga = (unsigned long long)(uintptr_t)gbase;   // byte address
  unsigned long long d0s = (unsigned long long)cin;               // dim0 stride (elems)
  unsigned long long d1s = (unsigned long long)WP * (unsigned)cin;// dim1 stride (elems)
  u32x4 g0;
  g0[0] = 1u;                                   // count=1, user mode
  g0[1] = lds_addr;                             // LDS byte address
  g0[2] = (unsigned)ga;                         // global_addr[31:0]
  g0[3] = (unsigned)((ga >> 32) & 0x01FFFFFFu) | (2u << 30);  // addr[56:32], type=2
  i32x8 g1;
  g1[0] = (int)(1u << 16);                                     // data_size = 2B
  g1[1] = (int)(((unsigned)cin & 0xFFFFu) << 16);              // tensor_dim0 lo16
  g1[2] = (int)(((unsigned)cin >> 16) | ((unsigned)WP << 16)); // dim0 hi16 | dim1 lo16
  g1[3] = (int)(32u << 16);                                    // dim1 hi16=0 | tile_dim0=32
  g1[4] = (int)((unsigned)TCOLS | ((unsigned)TROWS << 16));    // tile_dim1 | tile_dim2
  g1[5] = (int)(unsigned)(d0s & 0xFFFFFFFFull);                // dim0_stride[31:0]
  g1[6] = (int)((unsigned)((d0s >> 32) & 0xFFFFull) |
                (unsigned)((d1s & 0xFFFFull) << 16));          // d0s[47:32] | d1s[15:0]
  g1[7] = (int)(unsigned)((d1s >> 16) & 0xFFFFFFFFull);        // d1s[47:16]
  i32x4 g2; g2[0] = HP; g2[1] = 0; g2[2] = 0; g2[3] = 0;       // tensor_dim2 = 98
  i32x4 g3; g3[0] = 0; g3[1] = 0; g3[2] = 0; g3[3] = 0;
#if __has_include(<hip/amd_detail/amd_gfx1250_TDM.h>)
  i32x8 g4; g4[0]=0; g4[1]=0; g4[2]=0; g4[3]=0; g4[4]=0; g4[5]=0; g4[6]=0; g4[7]=0;
  __builtin_amdgcn_tensor_load_to_lds(g0, g1, g2, g3, g4, 0);  // clang-23 form
#else
  __builtin_amdgcn_tensor_load_to_lds(g0, g1, g2, g3, 0);      // ROCm 7.2 form
#endif
}

// ---------------- 3x3 conv as 9 shifted GEMMs via v_wmma_f32_16x16x32_f16 -----
// block = 128 thr = 4 waves; wave w -> Cout tile w (64 = 4x16).
// Block output tile: 1 row x 48 cols (3 N-tiles per wave, A reused 3x).
// Per 32-channel K chunk: wave0 TDM-DMAs the 3x50x32 f16 halo into LDS
// (double-buffered, overlapped with compute); waves build B fragments via ds_load_b128.
template <int CIN>
__global__ __launch_bounds__(128) void k_conv3x3_wmma(
    const _Float16* __restrict__ act,    // padded NHWC [98][98][CIN] f16
    const _Float16* __restrict__ wpack,  // packed A fragments (kc outer)
    const float* __restrict__ bias, const float* __restrict__ alpha,
    _Float16* __restrict__ out)          // padded NHWC [98][98][64] f16
{
  constexpr int KC = CIN / 32;
  __shared__ __align__(16) _Float16 smem[2][TROWS * TCOLS * 32];   // 2 x 9600 B

  const int lane  = threadIdx.x & 31;
  const int mtile = threadIdx.x >> 5;          // 0..3
  const int y  = blockIdx.x >> 1;              // output row 0..95
  const int x0 = (blockIdx.x & 1) * (NT * 16); // output col base 0 / 48
  const int col   = lane & 15;
  const int khalf = lane >> 4;

  v8f acc[NT];
  #pragma unroll
  for (int nt = 0; nt < NT; ++nt) acc[nt] = (v8f){};

  const _Float16* a0 = wpack + ((size_t)mtile * 32 + lane) * 16;
  const _Float16* gtile = act + ((size_t)y * WP + x0) * CIN;   // + kc*32 per chunk
  const unsigned ldsb0 = (unsigned)(uintptr_t)&smem[0][0];
  const unsigned ldsb1 = (unsigned)(uintptr_t)&smem[1][0];
  const bool wave0 = (threadIdx.x < 32);       // wave-uniform; TDM ignores EXEC

  if (wave0) tdm_load_tile(gtile, ldsb0, CIN);

  for (int kc = 0; kc < KC; ++kc) {
    if (wave0) __builtin_amdgcn_s_wait_tensorcnt(0);   // buf[kc&1] landed in LDS
    __syncthreads();   // orders TDM data for all waves; prior reads of other buf done
    if (wave0 && (kc + 1 < KC))
      tdm_load_tile(gtile + (kc + 1) * 32, (kc & 1) ? ldsb0 : ldsb1, CIN);

    const _Float16* bufc = &smem[kc & 1][0];
    const _Float16* akc  = a0 + (size_t)kc * 9 * 4 * 32 * 16;
    #pragma unroll
    for (int tap = 0; tap < 9; ++tap) {
      const int dy = tap / 3, dx = tap % 3;
      v16h a = *(const v16h*)(akc + (size_t)tap * 4 * 32 * 16);
      v16h b[NT];
      #pragma unroll
      for (int nt = 0; nt < NT; ++nt)
        b[nt] = *(const v16h*)&bufc[((dy * TCOLS) + nt * 16 + col + dx) * 32 + khalf * 16];
      #pragma unroll
      for (int nt = 0; nt < NT; ++nt)
        acc[nt] = __builtin_amdgcn_wmma_f32_16x16x32_f16(false, a, false, b[nt],
                                                         (short)0, acc[nt], false, false);
    }
  }

  const float al = alpha[0];
  #pragma unroll
  for (int nt = 0; nt < NT; ++nt) {
    #pragma unroll
    for (int v = 0; v < 8; ++v) {
      const int c = mtile * 16 + v + (khalf ? 8 : 0);  // C/D layout: M = v (+8 hi lanes)
      float r = acc[nt][v] + bias[c];
      r = (r >= 0.f) ? r : al * r;
      out[((size_t)(y + 1) * WP + (x0 + nt * 16 + col + 1)) * 64 + c] = (_Float16)r;
    }
  }
}

// ---------------- conv5 (64->1), f32 output ----------------
__global__ void k_conv5(const _Float16* __restrict__ act, const float* __restrict__ W5,
                        const float* __restrict__ b5, float* __restrict__ outp) {
  int t = blockIdx.x * blockDim.x + threadIdx.x;  // 96*96
  if (t >= H * W) return;
  int x = t % W, y = t / W;
  float acc = b5[0];
  for (int tap = 0; tap < 9; ++tap) {
    int dy = tap / 3, dx = tap % 3;
    const _Float16* p = act + ((size_t)(y + dy) * WP + (x + dx)) * 64;
    #pragma unroll 8
    for (int c = 0; c < 64; ++c) acc += W5[c * 9 + tap] * (float)p[c];
  }
  outp[t] = acc;
}

extern "C" void kernel_launch(void* const* d_in, const int* in_sizes, int n_in,
                              void* d_out, int out_size, void* d_ws, size_t ws_size,
                              hipStream_t stream) {
  const float* x   = (const float*)d_in[0];
  const float* Ws  = (const float*)d_in[1];
  const float* Wup = (const float*)d_in[2];
  const float* bup = (const float*)d_in[3];
  const float* Wc1 = (const float*)d_in[4];
  const float* bc1 = (const float*)d_in[5];
  const float* ac1 = (const float*)d_in[6];
  const float* W2  = (const float*)d_in[7];
  const float* b2  = (const float*)d_in[8];
  const float* a2  = (const float*)d_in[9];
  const float* W3  = (const float*)d_in[10];
  const float* b3  = (const float*)d_in[11];
  const float* a3  = (const float*)d_in[12];
  const float* W4  = (const float*)d_in[13];
  const float* b4  = (const float*)d_in[14];
  const float* a4  = (const float*)d_in[15];
  const float* W5  = (const float*)d_in[16];
  const float* b5  = (const float*)d_in[17];

  char* ws = (char*)d_ws;
  float*     s     = (float*)(ws + OFF_S);
  float*     psPad = (float*)(ws + OFF_PS);
  _Float16*  fused = (_Float16*)(ws + OFF_FUSED);
  _Float16*  actA  = (_Float16*)(ws + OFF_ACTA);
  _Float16*  actB  = (_Float16*)(ws + OFF_ACTB);
  _Float16*  actC  = (_Float16*)(ws + OFF_ACTC);
  _Float16*  w2p   = (_Float16*)(ws + OFF_W2P);
  _Float16*  w3p   = (_Float16*)(ws + OFF_W3P);
  _Float16*  w4p   = (_Float16*)(ws + OFF_W4P);

  // zero ps/fused/act buffers (padding borders)
  k_zero<<<2048, 256, 0, stream>>>((uint4*)ws, OFF_W2P / 16);

  k_sampling<<<(G * SC * 9 + 255) / 256, 256, 0, stream>>>(x, Ws, s);
  k_head<<<(G * H * W + 255) / 256, 256, 0, stream>>>(s, Wup, bup, psPad);
  k_conv1_fuse<<<(H * W * OUTC + 255) / 256, 256, 0, stream>>>(psPad, Wc1, bc1, ac1, fused);

  k_pack_w<<<(32 * 9 * 4 * 32 + 255) / 256, 256, 0, stream>>>(W2, w2p, 1024);
  k_pack_w<<<(2 * 9 * 4 * 32 + 255) / 256, 256, 0, stream>>>(W3, w3p, 64);
  k_pack_w<<<(2 * 9 * 4 * 32 + 255) / 256, 256, 0, stream>>>(W4, w4p, 64);

  const int nblk = H * (W / (NT * 16));  // 96 rows x 2 col-blocks = 192
  k_conv3x3_wmma<1024><<<nblk, 128, 0, stream>>>(fused, w2p, b2, a2, actA);
  k_conv3x3_wmma<64>  <<<nblk, 128, 0, stream>>>(actA,  w3p, b3, a3, actB);
  k_conv3x3_wmma<64>  <<<nblk, 128, 0, stream>>>(actB,  w4p, b4, a4, actC);

  k_conv5<<<(H * W + 255) / 256, 256, 0, stream>>>(actC, W5, b5, (float*)d_out);
}
